// Net_17686675325164
// MI455X (gfx1250) — compile-verified, matrix operand order checked by reading, not run on hardware
//
#include <hip/hip_runtime.h>
#include <hip/hip_bf16.h>

// ---------------------------------------------------------------------------
// Types for WMMA fragments (CDNA5 / gfx1250, wave32)
// ---------------------------------------------------------------------------
typedef _Float16 half16 __attribute__((ext_vector_type(16)));
typedef _Float16 half8  __attribute__((ext_vector_type(8)));
typedef float    floatx8 __attribute__((ext_vector_type(8)));

#define BATCH   1024
#define TSTEPS  32
#define HID     128
#define FEAT    4608
#define KTOT    4736          // FEAT + HID
#define NGATES  512           // 4*HID
#define MSKN    900
#define ACTN    26
#define ACT_TOTAL (BATCH * TSTEPS * ACTN)   // floats before msks in d_out

// A/B fragment: two 16-byte halves per CDNA5 16-bit 16x32 lane layout
// (lanes 0-15 hold K {0..7,16..23}, lanes 16-31 hold K {8..15,24..31};
//  the +kLo per-lane offset is folded into the base pointer).
struct Frag { half8 lo; half8 hi; };

__device__ __forceinline__ Frag load_frag(const _Float16* p) {
    Frag f;
    f.lo = *(const half8*)(p);
    f.hi = *(const half8*)(p + 16);
    return f;
}
__device__ __forceinline__ half16 frag_cast(const Frag& f) {
    return __builtin_bit_cast(half16, f);
}
__device__ __forceinline__ float sigmoidf_(float x) {
    return 1.0f / (1.0f + __expf(-x));
}

// ---------------------------------------------------------------------------
// One-time: concat + convert weights [w_ih | w_hh] -> f16 [512, 4736]
// ---------------------------------------------------------------------------
__global__ void prep_weights(const float* __restrict__ w_ih,
                             const float* __restrict__ w_hh,
                             _Float16* __restrict__ wcat) {
    int i = blockIdx.x * blockDim.x + threadIdx.x;
    if (i >= NGATES * KTOT) return;
    int n = i / KTOT, k = i % KTOT;
    float v = (k < FEAT) ? w_ih[n * FEAT + k] : w_hh[n * HID + (k - FEAT)];
    wcat[i] = (_Float16)v;
}

// ---------------------------------------------------------------------------
// One-time: c = 0, h-part of xh = 0 (h itself is written before first read)
// ---------------------------------------------------------------------------
__global__ void init_state(float* __restrict__ c, _Float16* xh) {
    int tid = blockIdx.x * blockDim.x + threadIdx.x;
    if (tid >= BATCH * HID) return;
    c[tid] = 0.0f;
    int b = tid >> 7, j = tid & 127;
    xh[(size_t)b * KTOT + FEAT + j] = (_Float16)0.0f;
}

// ---------------------------------------------------------------------------
// Conv chain body (shared between the t=0 kernel and the fused heads+conv
// kernel). s_in must be populated and synced before the call. Writes f16
// features into xh[b, 0:4608] (NCHW flatten: c*576 + y*24 + x).
// ---------------------------------------------------------------------------
__device__ __forceinline__ void conv_chain_dev(
    int b, int tid,
    const float* s_in, float* s_b1, float* s_b2,
    const float* s_w1, const float* s_w2, const float* s_w3,
    const float* c1b, const float* c2b, const float* c3b,
    _Float16* xh) {
    // conv1: 1->8 ch, 30x30 -> 28x28
    for (int idx = tid; idx < 8 * 784; idx += 256) {
        int oc = idx / 784, rem = idx % 784, oy = rem / 28, ox = rem % 28;
        float sum = c1b[oc];
        #pragma unroll
        for (int ky = 0; ky < 3; ++ky)
            #pragma unroll
            for (int kx = 0; kx < 3; ++kx)
                sum += s_in[(oy + ky) * 30 + ox + kx] * s_w1[oc * 9 + ky * 3 + kx];
        s_b1[idx] = sum > 0.0f ? sum : 0.0f;
    }
    __syncthreads();
    // conv2: 8->8 ch, 28x28 -> 26x26
    for (int idx = tid; idx < 8 * 676; idx += 256) {
        int oc = idx / 676, rem = idx % 676, oy = rem / 26, ox = rem % 26;
        float sum = c2b[oc];
        for (int ic = 0; ic < 8; ++ic) {
            const float* wp = &s_w2[oc * 72 + ic * 9];
            const float* ip = &s_b1[ic * 784 + oy * 28 + ox];
            #pragma unroll
            for (int ky = 0; ky < 3; ++ky)
                #pragma unroll
                for (int kx = 0; kx < 3; ++kx)
                    sum += ip[ky * 28 + kx] * wp[ky * 3 + kx];
        }
        s_b2[idx] = sum > 0.0f ? sum : 0.0f;
    }
    __syncthreads();
    // conv3: 8->8 ch, 26x26 -> 24x24 -> f16 to xh
    for (int idx = tid; idx < FEAT; idx += 256) {
        int oc = idx / 576, rem = idx % 576, oy = rem / 24, ox = rem % 24;
        float sum = c3b[oc];
        for (int ic = 0; ic < 8; ++ic) {
            const float* wp = &s_w3[oc * 72 + ic * 9];
            const float* ip = &s_b2[ic * 676 + oy * 26 + ox];
            #pragma unroll
            for (int ky = 0; ky < 3; ++ky)
                #pragma unroll
                for (int kx = 0; kx < 3; ++kx)
                    sum += ip[ky * 26 + kx] * wp[ky * 3 + kx];
        }
        sum = sum > 0.0f ? sum : 0.0f;
        xh[(size_t)b * KTOT + idx] = (_Float16)sum;
    }
}

// ---------------------------------------------------------------------------
// t=0 conv: mask is all-ones, so s_in = batch row directly.
// ---------------------------------------------------------------------------
__global__ void __launch_bounds__(256)
conv_first(const float* __restrict__ batch,
           const float* __restrict__ c1w, const float* __restrict__ c1b,
           const float* __restrict__ c2w, const float* __restrict__ c2b,
           const float* __restrict__ c3w, const float* __restrict__ c3b,
           _Float16* xh) {
    __shared__ float s_in[900];
    __shared__ float s_b1[8 * 784];
    __shared__ float s_b2[8 * 676];
    __shared__ float s_w1[72];
    __shared__ float s_w2[576];
    __shared__ float s_w3[576];
    const int b = blockIdx.x, tid = threadIdx.x;
    for (int i = tid; i < 900; i += 256) s_in[i] = batch[(size_t)b * 900 + i];
    for (int i = tid; i < 72; i += 256) s_w1[i] = c1w[i];
    for (int i = tid; i < 576; i += 256) { s_w2[i] = c2w[i]; s_w3[i] = c3w[i]; }
    __syncthreads();
    conv_chain_dev(b, tid, s_in, s_b1, s_b2, s_w1, s_w2, s_w3, c1b, c2b, c3b, xh);
}

// ---------------------------------------------------------------------------
// Fused WMMA GEMM + LSTM cell.
// Grid: 64 blocks (one 16-row M-tile each) x 256 threads (8 waves).
// Wave w covers N columns [w*64, w*64+64): 16x64 strip, register
// double-buffered K pipeline (10 b128 loads in flight over 4 WMMAs) so the
// compiler can issue partial s_wait_loadcnt instead of draining per WMMA.
// Whole 16x512 gate tile lands in LDS; the same block then runs the LSTM
// pointwise update for its 16 batch rows and emits h (f32 + f16-into-xh).
// ---------------------------------------------------------------------------
__global__ void __launch_bounds__(256)
gemm_lstm(_Float16* xh,                       // A rows + h-part feedback (aliased on purpose)
          const _Float16* __restrict__ wcat,
          const float* __restrict__ b_ih, const float* __restrict__ b_hh,
          float* __restrict__ h, float* __restrict__ c) {
    __shared__ float s_gates[16][NGATES];     // 32 KB

    const int lane = threadIdx.x & 31;
    const int wave = threadIdx.x >> 5;
    const int m0 = blockIdx.x << 4;
    const int n0 = wave << 6;
    const int l16 = lane & 15;
    const int kLo = (lane >> 4) << 3;         // 0 or 8

    const _Float16* arow  = xh   + (size_t)(m0 + l16) * KTOT + kLo;
    const _Float16* brow0 = wcat + (size_t)(n0 + l16) * KTOT + kLo;
    const _Float16* brow1 = brow0 + (size_t)16 * KTOT;
    const _Float16* brow2 = brow0 + (size_t)32 * KTOT;
    const _Float16* brow3 = brow0 + (size_t)48 * KTOT;

    floatx8 acc0 = {}, acc1 = {}, acc2 = {}, acc3 = {};

    // pipeline prologue: K-slab 0 in flight
    Frag fa = load_frag(arow);
    Frag f0 = load_frag(brow0);
    Frag f1 = load_frag(brow1);
    Frag f2 = load_frag(brow2);
    Frag f3 = load_frag(brow3);

    for (int k0 = 32; k0 < KTOT; k0 += 32) {
        Frag na  = load_frag(arow  + k0);
        Frag nb0 = load_frag(brow0 + k0);
        Frag nb1 = load_frag(brow1 + k0);
        Frag nb2 = load_frag(brow2 + k0);
        Frag nb3 = load_frag(brow3 + k0);
        half16 a = frag_cast(fa);
        acc0 = __builtin_amdgcn_wmma_f32_16x16x32_f16(false, a, false, frag_cast(f0), (short)0, acc0, false, false);
        acc1 = __builtin_amdgcn_wmma_f32_16x16x32_f16(false, a, false, frag_cast(f1), (short)0, acc1, false, false);
        acc2 = __builtin_amdgcn_wmma_f32_16x16x32_f16(false, a, false, frag_cast(f2), (short)0, acc2, false, false);
        acc3 = __builtin_amdgcn_wmma_f32_16x16x32_f16(false, a, false, frag_cast(f3), (short)0, acc3, false, false);
        fa = na; f0 = nb0; f1 = nb1; f2 = nb2; f3 = nb3;
    }
    {
        half16 a = frag_cast(fa);
        acc0 = __builtin_amdgcn_wmma_f32_16x16x32_f16(false, a, false, frag_cast(f0), (short)0, acc0, false, false);
        acc1 = __builtin_amdgcn_wmma_f32_16x16x32_f16(false, a, false, frag_cast(f1), (short)0, acc1, false, false);
        acc2 = __builtin_amdgcn_wmma_f32_16x16x32_f16(false, a, false, frag_cast(f2), (short)0, acc2, false, false);
        acc3 = __builtin_amdgcn_wmma_f32_16x16x32_f16(false, a, false, frag_cast(f3), (short)0, acc3, false, false);
    }

    // C/D layout: VGPR r -> M = mRel + r, mRel = 8*(lane>=16); N = n0+s*16+l16
    const int mRel = (lane >> 4) << 3;
    #pragma unroll
    for (int r = 0; r < 8; ++r) {
        s_gates[mRel + r][n0 + l16]      = acc0[r];
        s_gates[mRel + r][n0 + 16 + l16] = acc1[r];
        s_gates[mRel + r][n0 + 32 + l16] = acc2[r];
        s_gates[mRel + r][n0 + 48 + l16] = acc3[r];
    }
    __syncthreads();

    // LSTM pointwise for this block's 16 batch rows
    for (int idx = threadIdx.x; idx < 16 * HID; idx += 256) {
        int r = idx >> 7, j = idx & 127;
        int m = m0 + r;
        float gi = s_gates[r][j]           + b_ih[j]           + b_hh[j];
        float gf = s_gates[r][HID + j]     + b_ih[HID + j]     + b_hh[HID + j];
        float gg = s_gates[r][2 * HID + j] + b_ih[2 * HID + j] + b_hh[2 * HID + j];
        float go = s_gates[r][3 * HID + j] + b_ih[3 * HID + j] + b_hh[3 * HID + j];
        float cn = sigmoidf_(gf) * c[(size_t)m * HID + j] + sigmoidf_(gi) * tanhf(gg);
        float hn = sigmoidf_(go) * tanhf(cn);
        c[(size_t)m * HID + j] = cn;
        h[(size_t)m * HID + j] = hn;
        xh[(size_t)m * KTOT + FEAT + j] = (_Float16)hn;   // feedback for t+1
    }
}

// ---------------------------------------------------------------------------
// Fused heads + next-step conv. One block per batch row:
//   msk = sigmoid(h @ msk_w^T + msk_b) -> d_out (and LDS)
//   act = h @ act_w^T + act_b          -> d_out
//   if not last step: s_in = batch * msk, run conv chain -> xh[b, 0:4608]
// ---------------------------------------------------------------------------
__global__ void __launch_bounds__(256)
heads_conv(const float* __restrict__ h,
           const float* __restrict__ msk_w, const float* __restrict__ msk_b,
           const float* __restrict__ act_w, const float* __restrict__ act_b,
           const float* __restrict__ batch,
           const float* __restrict__ c1w, const float* __restrict__ c1b,
           const float* __restrict__ c2w, const float* __restrict__ c2b,
           const float* __restrict__ c3w, const float* __restrict__ c3b,
           _Float16* xh, float* __restrict__ out, int t, int do_conv) {
    __shared__ float s_h[HID];
    __shared__ float s_msk[MSKN];
    __shared__ float s_in[900];
    __shared__ float s_b1[8 * 784];
    __shared__ float s_b2[8 * 676];
    __shared__ float s_w1[72];
    __shared__ float s_w2[576];
    __shared__ float s_w3[576];

    const int b = blockIdx.x, tid = threadIdx.x;

    for (int i = tid; i < HID; i += 256) s_h[i] = h[(size_t)b * HID + i];
    if (do_conv) {
        for (int i = tid; i < 72; i += 256) s_w1[i] = c1w[i];
        for (int i = tid; i < 576; i += 256) { s_w2[i] = c2w[i]; s_w3[i] = c3w[i]; }
    }
    __syncthreads();

    for (int j = tid; j < MSKN + ACTN; j += 256) {
        if (j < MSKN) {
            const float* w = msk_w + (size_t)j * HID;
            float s = msk_b[j];
            #pragma unroll 8
            for (int k = 0; k < HID; ++k) s += s_h[k] * w[k];
            float v = sigmoidf_(s);
            s_msk[j] = v;
            out[ACT_TOTAL + (size_t)b * (TSTEPS * MSKN) + (size_t)t * MSKN + j] = v;
        } else {
            int ja = j - MSKN;
            const float* w = act_w + (size_t)ja * HID;
            float s = act_b[ja];
            #pragma unroll 8
            for (int k = 0; k < HID; ++k) s += s_h[k] * w[k];
            out[(size_t)b * (TSTEPS * ACTN) + (size_t)t * ACTN + ja] = s;
        }
    }

    if (do_conv) {   // uniform branch (kernel arg) -> barriers are safe
        __syncthreads();
        for (int i = tid; i < 900; i += 256)
            s_in[i] = batch[(size_t)b * 900 + i] * s_msk[i];
        __syncthreads();
        conv_chain_dev(b, tid, s_in, s_b1, s_b2, s_w1, s_w2, s_w3,
                       c1b, c2b, c3b, xh);
    }
}

// ---------------------------------------------------------------------------
// Host-side orchestration (graph-capture safe: launches only)
// ---------------------------------------------------------------------------
static inline size_t align256(size_t x) { return (x + 255) & ~(size_t)255; }

extern "C" void kernel_launch(void* const* d_in, const int* in_sizes, int n_in,
                              void* d_out, int out_size, void* d_ws, size_t ws_size,
                              hipStream_t stream) {
    const float* batch = (const float*)d_in[0];
    const float* c1w = (const float*)d_in[1];
    const float* c1b = (const float*)d_in[2];
    const float* c2w = (const float*)d_in[3];
    const float* c2b = (const float*)d_in[4];
    const float* c3w = (const float*)d_in[5];
    const float* c3b = (const float*)d_in[6];
    const float* w_ih = (const float*)d_in[7];
    const float* w_hh = (const float*)d_in[8];
    const float* b_ih = (const float*)d_in[9];
    const float* b_hh = (const float*)d_in[10];
    const float* msk_w = (const float*)d_in[11];
    const float* msk_b = (const float*)d_in[12];
    const float* act_w = (const float*)d_in[13];
    const float* act_b = (const float*)d_in[14];
    float* out = (float*)d_out;

    // Workspace carve-up
    char* ws = (char*)d_ws;
    size_t o = 0;
    _Float16* wcat = (_Float16*)(ws + o); o += align256((size_t)NGATES * KTOT * 2);
    _Float16* xh   = (_Float16*)(ws + o); o += align256((size_t)BATCH * KTOT * 2);
    float* h       = (float*)(ws + o);    o += align256((size_t)BATCH * HID * 4);
    float* c       = (float*)(ws + o);    o += align256((size_t)BATCH * HID * 4);
    (void)ws_size;

    {
        int n = NGATES * KTOT;
        prep_weights<<<(n + 255) / 256, 256, 0, stream>>>(w_ih, w_hh, wcat);
    }
    {
        int n = BATCH * HID;
        init_state<<<(n + 255) / 256, 256, 0, stream>>>(c, xh);
    }
    conv_first<<<BATCH, 256, 0, stream>>>(batch, c1w, c1b, c2w, c2b, c3w, c3b, xh);

    for (int t = 0; t < TSTEPS; ++t) {
        gemm_lstm<<<64, 256, 0, stream>>>(xh, wcat, b_ih, b_hh, h, c);
        heads_conv<<<BATCH, 256, 0, stream>>>(h, msk_w, msk_b, act_w, act_b,
                                              batch, c1w, c1b, c2w, c2b, c3w, c3b,
                                              xh, out, t, (t < TSTEPS - 1) ? 1 : 0);
    }
}
